// _DeformConvBase_28905129902135
// MI455X (gfx1250) — matrix operand order repaired and checked by reading, not compiled, not run on hardware
//
#include <hip/hip_runtime.h>
#include <hip/hip_bf16.h>

typedef __attribute__((ext_vector_type(16))) _Float16 v16h;
typedef __attribute__((ext_vector_type(8)))  float    v8f;

#define Bv    8
#define CIN   64
#define Hv    128
#define Wv    128
#define HWv   (Hv*Wv)          /* 16384 */
#define COUT  64
#define KTAPS 9
#define OFFC  18               /* 2*KTAPS offset channels */

#define OFF_ELEMS       ((size_t)Bv*OFFC*HWv)     /* 2,359,296 floats */
#define APACK_OFF_BYTES (OFF_ELEMS*4)             /* 9,437,184 B (32B aligned) */
#define APACK_FRAGS     (18*4*32)                 /* chunks * mblocks * lanes */
#define APACK_BYTES     (APACK_FRAGS*32)          /* 73,728 B */

// ---------------------------------------------------------------------------
// Kernel 1: offset convolution  (B,18,H,W) = conv3x3(x, w_offset) + b_offset
// One thread per output pixel computes all 18 channels; w_offset staged in LDS.
// ---------------------------------------------------------------------------
__global__ void __launch_bounds__(256)
dcn_offset_conv(const float* __restrict__ x,
                const float* __restrict__ w_off,
                const float* __restrict__ b_off,
                float* __restrict__ off_out) {
    __shared__ float sw[OFFC * CIN * 9];   // 41,472 B of 320 KB WGP LDS
    int tid = threadIdx.x;
    for (int i = tid; i < OFFC * CIN * 9; i += 256) sw[i] = w_off[i];
    __syncthreads();

    int pix = blockIdx.x * 256 + tid;      // 131072 pixels total
    int b   = pix >> 14;
    int rem = pix & (HWv - 1);
    int h   = rem >> 7;
    int w   = rem & (Wv - 1);

    float acc[OFFC];
#pragma unroll
    for (int oc = 0; oc < OFFC; ++oc) acc[oc] = b_off[oc];

    const float* xb = x + (size_t)b * CIN * HWv;
    for (int c = 0; c < CIN; ++c) {
#pragma unroll
        for (int i = 0; i < 3; ++i) {
            int y = h + i - 1;
#pragma unroll
            for (int j = 0; j < 3; ++j) {
                int xc = w + j - 1;
                float xv = (y >= 0 && y < Hv && xc >= 0 && xc < Wv)
                               ? xb[c * HWv + y * Wv + xc] : 0.0f;
                int t = c * 9 + i * 3 + j;   // uniform across wave -> LDS broadcast
#pragma unroll
                for (int oc = 0; oc < OFFC; ++oc)
                    acc[oc] = fmaf(xv, sw[oc * (CIN * 9) + t], acc[oc]);
            }
        }
    }
#pragma unroll
    for (int oc = 0; oc < OFFC; ++oc)
        off_out[(((size_t)b * OFFC + oc) * Hv + h) * Wv + w] = acc[oc];
}

// ---------------------------------------------------------------------------
// Kernel 2: repack w_conv (Cout,Cin,3,3) into WMMA A-fragment layout (f16).
// kd = tap*64 + c, chunks of 32. Fragment (chunk, mblock, lane) -> 16 halves.
// ISA A 16x32 f16 layout: row = lane&15; klo = (lane<16)?0:8;
//   halves h<8 -> K = klo+h ; h>=8 -> K = 16+klo+(h-8).
// ---------------------------------------------------------------------------
__global__ void dcn_pack_weights(const float* __restrict__ w_conv,
                                 _Float16* __restrict__ apack) {
    int t = blockIdx.x * 256 + threadIdx.x;
    if (t >= APACK_FRAGS) return;
    int lane  = t & 31;
    int mb    = (t >> 5) & 3;
    int chunk = t >> 7;            // 0..17
    int k     = chunk >> 1;        // tap
    int cb    = (chunk & 1) * 32;  // channel base of the 32-wide chunk
    int o     = mb * 16 + (lane & 15);
    int klo   = (lane < 16) ? 0 : 8;
#pragma unroll
    for (int h = 0; h < 16; ++h) {
        int kdl = (h < 8) ? (klo + h) : (16 + klo + (h - 8));
        int c   = cb + kdl;
        apack[(size_t)t * 16 + h] =
            (_Float16)w_conv[((size_t)o * CIN + c) * KTAPS + k];
    }
}

// ---------------------------------------------------------------------------
// Kernel 3: bilinear sample + WMMA GEMM + bias.
// Block = 256 threads = 8 waves; wave handles 16 pixels x 64 out channels.
// Grid = B*H = 1024 (one image row per block).
// A-fragments staged once per block into LDS with CDNA5 async-to-LDS copies
// (ASYNCcnt path), so the VMEM pipe is dedicated to the bilinear gathers.
// ---------------------------------------------------------------------------
__global__ void __launch_bounds__(256)
dcn_main(const float* __restrict__ x,
         const float* __restrict__ off,
         const _Float16* __restrict__ apack,
         const float* __restrict__ b_conv,
         float* __restrict__ out) {
    extern __shared__ char slds_raw[];            // APACK_BYTES (dynamic LDS)
    _Float16* slds = (_Float16*)slds_raw;

    int tid  = threadIdx.x;
    int lane = tid & 31;
    int wave = tid >> 5;                  // 0..7
    int row  = blockIdx.x;                // 0..1023
    int b    = row >> 7;
    int h    = row & 127;
    int w0   = wave * 16;
    int p    = lane & 15;                 // this lane's pixel column (B/C col)
    int wpix = w0 + p;
    int kadd = (lane >= 16) ? 8 : 0;      // B-fragment K-half per ISA layout

    // --- stage A-pack (72 KB) into LDS with async copies (GV addressing) ---
    {
        unsigned lds_base = (unsigned)(uintptr_t)slds;   // LDS byte offset
        const char* gbase = (const char*)apack;
#pragma unroll 1
        for (int i = tid; i < APACK_BYTES / 16; i += 256) {
            unsigned    lofs  = lds_base + (unsigned)i * 16u;
            const char* gaddr = gbase + (size_t)i * 16;
            asm volatile("global_load_async_to_lds_b128 %0, %1, off"
                         :: "v"(lofs), "v"(gaddr) : "memory");
        }
        asm volatile("s_wait_asynccnt 0" ::: "memory");
    }
    __syncthreads();

    v8f acc0 = {}, acc1 = {}, acc2 = {}, acc3 = {};

    const float* xb   = x + (size_t)b * CIN * HWv;
    const float* offb = off + (size_t)b * OFFC * HWv + (size_t)h * Wv + wpix;

    for (int k = 0; k < KTAPS; ++k) {
        // --- per-(pixel,tap) bilinear setup (reused across 32 channels per lane) ---
        float dy = offb[(size_t)(2 * k)     * HWv];
        float dx = offb[(size_t)(2 * k + 1) * HWv];
        float py = dy + (float)(k / 3) + (float)h    - 1.0f;
        float px = dx + (float)(k % 3) + (float)wpix - 1.0f;
        float y0f = floorf(py), x0f = floorf(px);
        float fy = py - y0f,   fx = px - x0f;
        float y1f = y0f + 1.0f, x1f = x0f + 1.0f;
        bool vy0 = (y0f >= 0.0f) && (y0f <= 127.0f);
        bool vy1 = (y1f >= 0.0f) && (y1f <= 127.0f);
        bool vx0 = (x0f >= 0.0f) && (x0f <= 127.0f);
        bool vx1 = (x1f >= 0.0f) && (x1f <= 127.0f);
        int yi0 = (int)fminf(fmaxf(y0f, 0.0f), 127.0f);
        int yi1 = (int)fminf(fmaxf(y1f, 0.0f), 127.0f);
        int xi0 = (int)fminf(fmaxf(x0f, 0.0f), 127.0f);
        int xi1 = (int)fminf(fmaxf(x1f, 0.0f), 127.0f);
        float w00 = (1.0f - fy) * (1.0f - fx) * ((vy0 && vx0) ? 1.0f : 0.0f);
        float w01 = (1.0f - fy) * fx          * ((vy0 && vx1) ? 1.0f : 0.0f);
        float w10 = fy * (1.0f - fx)          * ((vy1 && vx0) ? 1.0f : 0.0f);
        float w11 = fy * fx                   * ((vy1 && vx1) ? 1.0f : 0.0f);
        int i00 = yi0 * Wv + xi0, i01 = yi0 * Wv + xi1;
        int i10 = yi1 * Wv + xi0, i11 = yi1 * Wv + xi1;

#pragma unroll
        for (int cbi = 0; cbi < 2; ++cbi) {
            // Build B fragment: col = lane&15, K(h) = kadd + h (h<8), 16 + kadd + (h-8) (h>=8)
            const float* pl = xb + (size_t)(cbi * 32 + kadd) * HWv;
            v16h bfrag;
#pragma unroll
            for (int hh = 0; hh < 16; ++hh) {
                float v = w00 * pl[i00] + w01 * pl[i01]
                        + w10 * pl[i10] + w11 * pl[i11];
                bfrag[hh] = (_Float16)v;
                pl += (hh == 7) ? (size_t)9 * HWv : (size_t)HWv;
            }
            // A fragments from LDS: 4 blocks of 16 output channels
            const v16h* ap = (const v16h*)slds
                           + ((size_t)(k * 2 + cbi) * 4) * 32 + lane;
            acc0 = __builtin_amdgcn_wmma_f32_16x16x32_f16(
                       false, ap[0],  false, bfrag, (short)0, acc0, false, false);
            acc1 = __builtin_amdgcn_wmma_f32_16x16x32_f16(
                       false, ap[32], false, bfrag, (short)0, acc1, false, false);
            acc2 = __builtin_amdgcn_wmma_f32_16x16x32_f16(
                       false, ap[64], false, bfrag, (short)0, acc2, false, false);
            acc3 = __builtin_amdgcn_wmma_f32_16x16x32_f16(
                       false, ap[96], false, bfrag, (short)0, acc3, false, false);
        }
    }

    // C/D layout: VGPR i -> M = i + (lane>=16 ? 8 : 0), N = lane&15
    int obase = (lane >= 16) ? 8 : 0;
    float* outb = out + (size_t)b * COUT * HWv + (size_t)h * Wv + wpix;
#pragma unroll
    for (int i = 0; i < 8; ++i) {
        int o0 = obase + i;
        outb[(size_t)(o0)      * HWv] = acc0[i] + b_conv[o0];
        outb[(size_t)(o0 + 16) * HWv] = acc1[i] + b_conv[o0 + 16];
        outb[(size_t)(o0 + 32) * HWv] = acc2[i] + b_conv[o0 + 32];
        outb[(size_t)(o0 + 48) * HWv] = acc3[i] + b_conv[o0 + 48];
    }
}

// ---------------------------------------------------------------------------
extern "C" void kernel_launch(void* const* d_in, const int* in_sizes, int n_in,
                              void* d_out, int out_size, void* d_ws, size_t ws_size,
                              hipStream_t stream) {
    const float* x      = (const float*)d_in[0];
    const float* w_off  = (const float*)d_in[1];
    const float* b_off  = (const float*)d_in[2];
    const float* w_conv = (const float*)d_in[3];
    const float* b_conv = (const float*)d_in[4];
    float*       outp   = (float*)d_out;

    float*    off   = (float*)d_ws;                               // 9.44 MB
    _Float16* apack = (_Float16*)((char*)d_ws + APACK_OFF_BYTES); // +72 KB

    // 1) offset conv: 131072 pixels, 1 thread each
    dcn_offset_conv<<<512, 256, 0, stream>>>(x, w_off, b_off, off);
    // 2) repack conv weights into f16 WMMA A-fragments
    dcn_pack_weights<<<(APACK_FRAGS + 255) / 256, 256, 0, stream>>>(w_conv, apack);
    // 3) fused bilinear-sample + WMMA GEMM + bias (72 KB dynamic LDS)
    dcn_main<<<Bv * Hv, 256, APACK_BYTES, stream>>>(x, off, apack, b_conv, outp);
}
